// DMPNN_67104569032926
// MI455X (gfx1250) — matrix operand order repaired and checked by reading, not compiled, not run on hardware
//
#include <hip/hip_runtime.h>
#include <hip/hip_bf16.h>
#include <math.h>

#define NN   50000   // nodes
#define NE   800000  // edges
#define NG   2000    // graphs
#define FD   256     // input features
#define HD   256     // hidden
#define FPD  1034    // fingerprint
#define KCAT 1290    // HD + FPD
#define KPAD 1312    // KCAT padded to multiple of 32 (41*32)

typedef __attribute__((ext_vector_type(16))) _Float16 v16h;
typedef __attribute__((ext_vector_type(8)))  _Float16 v8h;
typedef __attribute__((ext_vector_type(8)))  float    v8f;

// ---------------------------------------------------------------------------
// f32 -> f16 elementwise convert
// ---------------------------------------------------------------------------
__global__ void cvt_f16_kernel(const float* __restrict__ in, _Float16* __restrict__ out, size_t n) {
    size_t i = (size_t)blockIdx.x * blockDim.x + threadIdx.x;
    if (i < n) out[i] = (_Float16)in[i];
}

// ---------------------------------------------------------------------------
// Weight transpose + convert: W (K x N, f32, row-major) -> Wt (N x Kp, f16),
// zero-padding K..Kp so the GEMM K-loop can run on multiples of 32.
// ---------------------------------------------------------------------------
__global__ void wt_cvt_kernel(const float* __restrict__ W, _Float16* __restrict__ Wt,
                              int K, int N, int Kp) {
    size_t idx = (size_t)blockIdx.x * blockDim.x + threadIdx.x;
    size_t total = (size_t)N * Kp;
    if (idx >= total) return;
    int n = (int)(idx / Kp);
    int k = (int)(idx % Kp);
    Wt[idx] = (k < K) ? (_Float16)W[(size_t)k * N + n] : (_Float16)0.0f;
}

// ---------------------------------------------------------------------------
// WMMA GEMM: C(MxN,f32) = act(A(MxK,f16) @ Bt^T + bias), Bt is N x K (f16).
// Block = 128 threads = 4 waves. Each wave owns a 32x64 tile (2 row-tiles x
// 4 col-tiles = 8 accumulators); the 4 B-frags per k-step are shared by both
// row-tiles, halving weight traffic vs a 16-row tile (24 B/wmma).
// Block covers 32 rows x 256 cols. K multiple of 32. act: 0=none,1=relu,2=tanh.
// If C16 != nullptr, also emits an f16 copy of the activated output (feeds the
// next GEMM without a separate convert pass).
// Row-tile 1 is uniformly clamped to M-16 when M % 32 == 16 so EXEC stays
// all-ones for WMMA; its store is guarded by a wave-uniform predicate.
// A-frag per ISA: lane group g=(lane>>4) holds K in {8g..8g+7} U {8g+16..8g+23}.
// B-frag: same K layout as A-frag, for column lane&15.
// ---------------------------------------------------------------------------
__global__ void __launch_bounds__(128)
gemm_wmma_kernel(const _Float16* __restrict__ A, const _Float16* __restrict__ Bt,
                 const float* __restrict__ bias, float* __restrict__ C,
                 _Float16* __restrict__ C16,
                 int M, int N, int K, int act) {
    const int lane     = threadIdx.x & 31;
    const int wave     = threadIdx.x >> 5;
    const int row0     = blockIdx.x * 32;
    const bool r1valid = (row0 + 16) < M;            // uniform across wave
    const int row1     = r1valid ? (row0 + 16) : (M - 16);
    const int col_base = blockIdx.y * 256 + wave * 64;
    const int ml       = lane & 15;                  // row in tile (A) / col in tile (B,C)
    const int grp      = lane >> 4;                  // lane group 0/1
    const int a_klo    = grp * 8;                    // A-frag low-K offset within 32-chunk
    const int b_k      = grp * 8;                    // B-frag K offset within 32-chunk

    const _Float16* arow0 = A + (size_t)(row0 + ml) * K;
    const _Float16* arow1 = A + (size_t)(row1 + ml) * K;

    v8f acc0[4] = {};
    v8f acc1[4] = {};

    for (int kt = 0; kt < K; kt += 32) {
        v8h a0lo = *(const v8h*)(arow0 + kt + a_klo);
        v8h a0hi = *(const v8h*)(arow0 + kt + a_klo + 16);
        v8h a1lo = *(const v8h*)(arow1 + kt + a_klo);
        v8h a1hi = *(const v8h*)(arow1 + kt + a_klo + 16);
        v16h a0 = __builtin_shufflevector(a0lo, a0hi, 0,1,2,3,4,5,6,7,8,9,10,11,12,13,14,15);
        v16h a1 = __builtin_shufflevector(a1lo, a1hi, 0,1,2,3,4,5,6,7,8,9,10,11,12,13,14,15);
#pragma unroll
        for (int j = 0; j < 4; ++j) {
            int n = col_base + j * 16 + ml;
            v8h blo = *(const v8h*)(Bt + (size_t)n * K + kt + b_k);
            v8h bhi = *(const v8h*)(Bt + (size_t)n * K + kt + b_k + 16);
            v16h b = __builtin_shufflevector(blo, bhi, 0,1,2,3,4,5,6,7,8,9,10,11,12,13,14,15);
            acc0[j] = __builtin_amdgcn_wmma_f32_16x16x32_f16(
                false, a0, false, b, (short)0, acc0[j], false, false);
            acc1[j] = __builtin_amdgcn_wmma_f32_16x16x32_f16(
                false, a1, false, b, (short)0, acc1[j], false, false);
        }
    }

    const int row_off = grp * 8;
#pragma unroll
    for (int j = 0; j < 4; ++j) {
        int col = col_base + j * 16 + ml;
        float bv = bias[col];
#pragma unroll
        for (int r = 0; r < 8; ++r) {
            float v = acc0[j][r] + bv;
            if (act == 1)      v = fmaxf(v, 0.0f);
            else if (act == 2) v = tanhf(v);
            size_t idx = (size_t)(row0 + row_off + r) * N + col;
            C[idx] = v;
            if (C16) C16[idx] = (_Float16)v;
        }
        if (r1valid) {
#pragma unroll
            for (int r = 0; r < 8; ++r) {
                float v = acc1[j][r] + bv;
                if (act == 1)      v = fmaxf(v, 0.0f);
                else if (act == 2) v = tanhf(v);
                size_t idx = (size_t)(row1 + row_off + r) * N + col;
                C[idx] = v;
                if (C16) C16[idx] = (_Float16)v;
            }
        }
    }
}

// ---------------------------------------------------------------------------
// aggr = relu(y)  (initializes the segment-sum with the self-loop message,
// since self-loop edge_attr == 0)
// ---------------------------------------------------------------------------
__global__ void relu_copy_kernel(const float* __restrict__ y, float* __restrict__ aggr, size_t n) {
    size_t i = (size_t)blockIdx.x * blockDim.x + threadIdx.x;
    if (i < n) aggr[i] = fmaxf(y[i], 0.0f);
}

// ---------------------------------------------------------------------------
// Edge scatter: aggr[dst] += relu(y[src] + ea * we).  4 edges per 256-thread
// block; 64 lanes x float4 per edge. y/aggr are L2-resident (51 MB each).
// ---------------------------------------------------------------------------
__global__ void edge_msg_kernel(const float* __restrict__ y,
                                const int* __restrict__ src, const int* __restrict__ dst,
                                const float* __restrict__ ea, const float* __restrict__ we,
                                float* __restrict__ aggr, int ne) {
    int t = threadIdx.x;
    int e = blockIdx.x * 4 + (t >> 6);
    if (e >= ne) return;
    int f4 = t & 63;
    int s = src[e];
    int d = dst[e];
    float a = ea[e];
    float4 yv = ((const float4*)(y + (size_t)s * HD))[f4];
    float4 wv = ((const float4*)we)[f4];
    float4 m;
    m.x = fmaxf(yv.x + a * wv.x, 0.0f);
    m.y = fmaxf(yv.y + a * wv.y, 0.0f);
    m.z = fmaxf(yv.z + a * wv.z, 0.0f);
    m.w = fmaxf(yv.w + a * wv.w, 0.0f);
    float* dp = aggr + (size_t)d * HD + f4 * 4;
    atomicAdd(dp + 0, m.x);
    atomicAdd(dp + 1, m.y);
    atomicAdd(dp + 2, m.z);
    atomicAdd(dp + 3, m.w);
}

__global__ void zero_f32_kernel(float* __restrict__ p, size_t n) {
    size_t i = (size_t)blockIdx.x * blockDim.x + threadIdx.x;
    if (i < n) p[i] = 0.0f;
}

// ---------------------------------------------------------------------------
// Per-node gate + pooled scatter: gate = sigmoid(dot(t1[i], g2w) + g2b);
// pooled[batch[i]] += x[i] * gate.  One block (256 threads) per node.
// ---------------------------------------------------------------------------
__global__ void gate_pool_kernel(const float* __restrict__ t1, const float* __restrict__ g2w,
                                 const float* __restrict__ g2b, const float* __restrict__ xf,
                                 const int* __restrict__ batch, float* __restrict__ pooled) {
    int i = blockIdx.x;
    int c = threadIdx.x;
    __shared__ float red[256];
    red[c] = t1[(size_t)i * HD + c] * g2w[c];
    __syncthreads();
    for (int s = 128; s > 0; s >>= 1) {
        if (c < s) red[c] += red[c + s];
        __syncthreads();
    }
    float gate = 1.0f / (1.0f + expf(-(red[0] + g2b[0])));
    int b = batch[i];
    atomicAdd(&pooled[(size_t)b * HD + c], xf[(size_t)i * HD + c] * gate);
}

// ---------------------------------------------------------------------------
// Build concatenated f16 input [pooled | fingerprint | 0-pad] (NG x KPAD)
// ---------------------------------------------------------------------------
__global__ void hcat_kernel(const float* __restrict__ pooled, const float* __restrict__ fp,
                            _Float16* __restrict__ hcat) {
    size_t idx = (size_t)blockIdx.x * blockDim.x + threadIdx.x;
    if (idx >= (size_t)NG * KPAD) return;
    int r = (int)(idx / KPAD);
    int c = (int)(idx % KPAD);
    float v = 0.0f;
    if (c < HD)         v = pooled[(size_t)r * HD + c];
    else if (c < KCAT)  v = fp[(size_t)r * FPD + (c - HD)];
    hcat[idx] = (_Float16)v;
}

// ---------------------------------------------------------------------------
// BatchNorm stats per column (one block per column, reduce over NG rows)
// ---------------------------------------------------------------------------
__global__ void bn_stats_kernel(const float* __restrict__ h, float* __restrict__ mu,
                                float* __restrict__ rsig, int rows) {
    int c = blockIdx.x;
    int t = threadIdx.x;
    float s = 0.0f, s2 = 0.0f;
    for (int r = t; r < rows; r += 256) {
        float v = h[(size_t)r * HD + c];
        s += v;
        s2 += v * v;
    }
    __shared__ float rs[256], rq[256];
    rs[t] = s; rq[t] = s2;
    __syncthreads();
    for (int k = 128; k > 0; k >>= 1) {
        if (t < k) { rs[t] += rs[t + k]; rq[t] += rq[t + k]; }
        __syncthreads();
    }
    if (t == 0) {
        float m = rs[0] / rows;
        mu[c] = m;
        rsig[c] = rsqrtf(rq[0] / rows - m * m + 1e-5f);
    }
}

// ---------------------------------------------------------------------------
// Fused BN + relu + fc2 reduction: one block per graph row
// ---------------------------------------------------------------------------
__global__ void final_kernel(const float* __restrict__ h, const float* __restrict__ mu,
                             const float* __restrict__ rsig, const float* __restrict__ gamma,
                             const float* __restrict__ beta, const float* __restrict__ fc2w,
                             const float* __restrict__ fc2b, float* __restrict__ out) {
    int r = blockIdx.x;
    int c = threadIdx.x;
    float v = (h[(size_t)r * HD + c] - mu[c]) * rsig[c] * gamma[c] + beta[c];
    v = fmaxf(v, 0.0f) * fc2w[c];
    __shared__ float red[256];
    red[c] = v;
    __syncthreads();
    for (int s = 128; s > 0; s >>= 1) {
        if (c < s) red[c] += red[c + s];
        __syncthreads();
    }
    if (c == 0) out[r] = red[0] + fc2b[0];
}

// ---------------------------------------------------------------------------
// kernel_launch
// ---------------------------------------------------------------------------
extern "C" void kernel_launch(void* const* d_in, const int* in_sizes, int n_in,
                              void* d_out, int out_size, void* d_ws, size_t ws_size,
                              hipStream_t stream) {
    const float* x     = (const float*)d_in[0];
    const int*   eidx  = (const int*)d_in[1];
    const float* eattr = (const float*)d_in[2];
    const float* fp    = (const float*)d_in[3];
    const int*   batch = (const int*)d_in[4];
    const float* lin_w = (const float*)d_in[5];   // (L, FD+1, HD)
    const float* lin_b = (const float*)d_in[6];   // (L, HD)
    const float* upd_w = (const float*)d_in[7];   // (L, HD, HD)
    const float* upd_b = (const float*)d_in[8];   // (L, HD)
    const float* g1_w  = (const float*)d_in[9];
    const float* g1_b  = (const float*)d_in[10];
    const float* g2_w  = (const float*)d_in[11];
    const float* g2_b  = (const float*)d_in[12];
    const float* fc1_w = (const float*)d_in[13];  // (KCAT, HD)
    const float* fc1_b = (const float*)d_in[14];
    const float* gamma = (const float*)d_in[15];
    const float* beta  = (const float*)d_in[16];
    const float* fc2_w = (const float*)d_in[17];
    const float* fc2_b = (const float*)d_in[18];
    float* out = (float*)d_out;

    // ---- workspace carve-out (256 B aligned) ----
    char* ws = (char*)d_ws;
    size_t off = 0;
    auto alloc = [&](size_t bytes) -> void* {
        void* p = ws + off;
        off = (off + bytes + 255) & ~(size_t)255;
        return p;
    };
    _Float16* a16x   = (_Float16*)alloc((size_t)NN * HD * sizeof(_Float16)); // activation f16 (GEMM A)
    _Float16* a16g   = (_Float16*)alloc((size_t)NN * HD * sizeof(_Float16)); // aggr f16 (GEMM A)
    float*    yb     = (float*)   alloc((size_t)NN * HD * sizeof(float));
    float*    aggr   = (float*)   alloc((size_t)NN * HD * sizeof(float));
    _Float16* wt     = (_Float16*)alloc((size_t)HD * KPAD * sizeof(_Float16));
    float*    pooled = (float*)   alloc((size_t)NG * HD * sizeof(float));
    _Float16* hcat   = (_Float16*)alloc((size_t)NG * KPAD * sizeof(_Float16));
    float*    hb     = (float*)   alloc((size_t)NG * HD * sizeof(float));
    float*    mu     = (float*)   alloc(HD * sizeof(float));
    float*    rsig   = (float*)   alloc(HD * sizeof(float));

    const size_t nodeElems  = (size_t)NN * HD;          // 12.8M
    const int    cvtBlocks  = (int)((nodeElems + 255) / 256);
    const int    gemmBlocksN = (NN + 31) / 32;          // 1563 (last block: 16 rows)
    const int    gemmBlocksG = (NG + 31) / 32;          // 63
    const int*   src = eidx;
    const int*   dst = eidx + NE;

    // first layer: convert input features once
    cvt_f16_kernel<<<cvtBlocks, 256, 0, stream>>>(x, a16x, nodeElems);

    for (int l = 0; l < 4; ++l) {
        const float* wn = lin_w + (size_t)l * (FD + 1) * HD;        // (FD, HD)
        const float* we = wn + (size_t)FD * HD;                     // (HD,)
        const float* lb = lin_b + (size_t)l * HD;
        const float* uw = upd_w + (size_t)l * HD * HD;
        const float* ub = upd_b + (size_t)l * HD;

        // y = x16 @ wn + lin_b  (f16 WMMA, f32 accumulate)
        wt_cvt_kernel<<<(HD * HD + 255) / 256, 256, 0, stream>>>(wn, wt, FD, HD, HD);
        gemm_wmma_kernel<<<dim3(gemmBlocksN, 1), 128, 0, stream>>>(
            a16x, wt, lb, yb, (_Float16*)nullptr, NN, HD, HD, 0);

        // aggr = relu(y)  (self loops), then += relu(y[src] + ea*we) over edges
        relu_copy_kernel<<<cvtBlocks, 256, 0, stream>>>(yb, aggr, nodeElems);
        edge_msg_kernel<<<NE / 4, 256, 0, stream>>>(yb, src, dst, eattr, we, aggr, NE);

        // x = relu(aggr @ upd_w + upd_b); also emit f16 copy into a16x for next GEMM
        cvt_f16_kernel<<<cvtBlocks, 256, 0, stream>>>(aggr, a16g, nodeElems);
        wt_cvt_kernel<<<(HD * HD + 255) / 256, 256, 0, stream>>>(uw, wt, HD, HD, HD);
        gemm_wmma_kernel<<<dim3(gemmBlocksN, 1), 128, 0, stream>>>(
            a16g, wt, ub, yb, a16x, NN, HD, HD, 1);
    }

    // t1 = tanh(x @ g1_w + g1_b)   (x f16 already in a16x, x f32 in yb)
    wt_cvt_kernel<<<(HD * HD + 255) / 256, 256, 0, stream>>>(g1_w, wt, HD, HD, HD);
    gemm_wmma_kernel<<<dim3(gemmBlocksN, 1), 128, 0, stream>>>(
        a16x, wt, g1_b, aggr, (_Float16*)nullptr, NN, HD, HD, 2);

    // pooled = segment_sum(x * sigmoid(t1 @ g2_w + g2_b), batch)
    zero_f32_kernel<<<(NG * HD + 255) / 256, 256, 0, stream>>>(pooled, (size_t)NG * HD);
    gate_pool_kernel<<<NN, 256, 0, stream>>>(aggr, g2_w, g2_b, yb, batch, pooled);

    // h = [pooled | fingerprint] @ fc1_w + fc1_b  (K padded 1290 -> 1312)
    hcat_kernel<<<(int)(((size_t)NG * KPAD + 255) / 256), 256, 0, stream>>>(pooled, fp, hcat);
    wt_cvt_kernel<<<(HD * KPAD + 255) / 256, 256, 0, stream>>>(fc1_w, wt, KCAT, HD, KPAD);
    gemm_wmma_kernel<<<dim3(gemmBlocksG, 1), 128, 0, stream>>>(
        hcat, wt, fc1_b, hb, (_Float16*)nullptr, NG, HD, KPAD, 0);

    // batchnorm + relu + fc2
    bn_stats_kernel<<<HD, 256, 0, stream>>>(hb, mu, rsig, NG);
    final_kernel<<<NG, 256, 0, stream>>>(hb, mu, rsig, gamma, beta, fc2_w, fc2_b, out);
}
